// VectorQuantizerEMA_54941221651334
// MI455X (gfx1250) — matrix operand order repaired and checked by reading, not compiled, not run on hardware
//
#include <hip/hip_runtime.h>
#include <hip/hip_bf16.h>

typedef __attribute__((ext_vector_type(4)))  float        v4f;
typedef __attribute__((ext_vector_type(8)))  float        v8f;
typedef __attribute__((ext_vector_type(8)))  __bf16       v8bf;
typedef __attribute__((ext_vector_type(16))) __bf16       v16bf;
typedef __attribute__((ext_vector_type(4)))  unsigned int u32x4;
typedef __attribute__((ext_vector_type(8)))  int          i32x8;
typedef __attribute__((ext_vector_type(4)))  int          i32x4;

#define N_ROWS   131072
#define N_EMB    1024
#define DIM      64
#define NTILES   (N_EMB / 16)   // 64 column tiles
#define CROW     136            // combined LDS tile row stride in bf16: [64 hi][64 lo][8 pad] = 272 B

#if __has_builtin(__builtin_amdgcn_tensor_load_to_lds) && __has_builtin(__builtin_amdgcn_s_wait_tensorcnt)
#define USE_TDM 1
#else
#define USE_TDM 0
#endif

// ws layout:
//   f32:  [0,1024) count_sum | [1024,66560) weight_sum | [66560,67584) enorm
//   bf16: embC [1024 rows x 128] = per-row [64 hi | 64 lo]  (256 KB, at float offset 67584)

// ---------------------------------------------------------------------------
// Kernel 0: zero accumulators, compute ||e_k||^2, split codebook into
// row-interleaved bf16 hi/lo (one 256 B row per code vector -> one TDM tile).
// ---------------------------------------------------------------------------
__global__ __launch_bounds__(256) void vq_init_kernel(const float* __restrict__ emb,
                                                      float* __restrict__ ws,
                                                      unsigned short* __restrict__ embC) {
    int i = blockIdx.x * 256 + threadIdx.x;
    if (i < N_EMB * DIM) {
        ws[N_EMB + i] = 0.0f;                  // weight_sum
        int r = i >> 6, d = i & 63;
        float x = emb[i];
        __bf16 h = (__bf16)x;
        __bf16 l = (__bf16)(x - (float)h);
        embC[r * 128 + d]      = __builtin_bit_cast(unsigned short, h);
        embC[r * 128 + 64 + d] = __builtin_bit_cast(unsigned short, l);
    }
    if (i < N_EMB) {
        ws[i] = 0.0f;                          // count_sum
        const float* e = emb + i * DIM;
        float s = 0.0f;
#pragma unroll
        for (int k = 0; k < DIM; ++k) s += e[k] * e[k];
        ws[N_EMB + N_EMB * DIM + i] = s;       // enorm
    }
}

#if USE_TDM
// Issue one TDM tile load: 16 codebook rows x 256 B (hi|lo) -> LDS, padded
// +16 B per 256 B row (272 B LDS row stride). data_size = 8 B units.
__device__ __forceinline__ void tdm_load_tile(const unsigned short* embC, int n0,
                                              unsigned lds_off) {
    unsigned long long ga = (unsigned long long)(uintptr_t)embC +
                            (unsigned long long)n0 * 256ull;
    u32x4 g0 = { 1u,                                   // count=1 (valid), user mode
                 lds_off,                              // LDS byte address
                 (unsigned)ga,                         // global_addr[31:0]
                 ((unsigned)(ga >> 32) & 0x01FFFFFFu) | 0x80000000u };  // [56:32] | type=2
    i32x8 g1 = { (int)((3u << 16) | (1u << 20) | (5u << 22) | (3u << 25)),
                 //    data_size=8B  pad_enable  pad_ivl=256B  pad_amt=16B
                 (int)(32u << 16),    // tensor_dim0 = 32 (8B units per row)
                 (int)(16u << 16),    // tensor_dim1 = 16 rows
                 (int)(32u << 16),    // tile_dim0 = 32
                 16,                  // tile_dim1 = 16, tile_dim2 = 0
                 32,                  // tensor_dim0_stride = 32 (contiguous 256B rows)
                 0, 0 };
    i32x4 gz4 = { 0, 0, 0, 0 };
    i32x8 gz8 = { 0, 0, 0, 0, 0, 0, 0, 0 };
    // 6-arg toolchain form: (g0, g1, g2, g3, extra, cpol)
    __builtin_amdgcn_tensor_load_to_lds(g0, g1, gz4, gz4, gz8, 0);
}
#endif

// ---------------------------------------------------------------------------
// Kernel 1: split-bf16 WMMA distance + argmin + gather + EMA accumulation.
// One wave per 16-row tile of z_e; 8 waves (256 threads) per block.
// dot(z,e) = hi_z*hi_e + hi_z*lo_e + lo_z*hi_e  (f32 accumulate) ~ f32 accuracy.
// Codebook tiles arrive via TDM (double-buffered, 2 tiles in flight).
// ---------------------------------------------------------------------------
__global__ __launch_bounds__(256) void vq_main_kernel(const float* __restrict__ z_e,
                                                      const float* __restrict__ emb,
                                                      const unsigned short* __restrict__ embC,
                                                      const float* __restrict__ enorm,
                                                      float* __restrict__ zq_out,
                                                      float* __restrict__ idx_out,
                                                      float* __restrict__ count_sum,
                                                      float* __restrict__ weight_sum) {
    __shared__ __align__(16) __bf16 BtC[2][16 * CROW];  // 2 x 4352 B tile buffers
    __shared__ __align__(16) float  senorm[N_EMB];      // 4 KB
    __shared__ int sidx[8 * 16];

    const int lane = threadIdx.x & 31;
    const int wave = threadIdx.x >> 5;
    const int m0   = (blockIdx.x * 8 + wave) * 16;  // first z_e row of this wave's tile
    const int row  = lane & 15;                     // M (A) / N (B,C) position
    const int kh   = lane >> 4;                     // K-half selector

    // Stage ||e||^2 into LDS once per block (256 threads x float4).
    *(v4f*)&senorm[threadIdx.x * 4] = *(const v4f*)&enorm[threadIdx.x * 4];

    // --- A fragments: z row split to bf16 hi/lo, two K=32 chunks.
    // 16-bit A layout: lane l holds M=l%16, K = c*32 + {8*kh..+7, 16+8*kh..+7}.
    v16bf a_hi[2], a_lo[2];
    {
        const float* zrow = z_e + (size_t)(m0 + row) * DIM;
#pragma unroll
        for (int c = 0; c < 2; ++c) {
#pragma unroll
            for (int j = 0; j < 8; ++j) {
                float x0 = zrow[c * 32 + 8 * kh + j];
                __bf16 h0 = (__bf16)x0;
                a_hi[c][j] = h0;
                a_lo[c][j] = (__bf16)(x0 - (float)h0);
                float x1 = zrow[c * 32 + 16 + 8 * kh + j];
                __bf16 h1 = (__bf16)x1;
                a_hi[c][8 + j] = h1;
                a_lo[c][8 + j] = (__bf16)(x1 - (float)h1);
            }
        }
    }

    float best[8];
    int   bidx[8];
#pragma unroll
    for (int v = 0; v < 8; ++v) { best[v] = 3.4e38f; bidx[v] = 0; }

#if USE_TDM
    // Prologue: launch tiles 0 and 1 into the two buffers (wave 0 drives the TDM).
    if (wave == 0) {
        tdm_load_tile(embC, 0,  (unsigned)(size_t)&BtC[0][0]);
        tdm_load_tile(embC, 16, (unsigned)(size_t)&BtC[1][0]);
    }
#endif

#pragma unroll 1
    for (int nt = 0; nt < NTILES; ++nt) {
        const int n0 = nt * 16;

#if USE_TDM
        if (wave == 0) {
            if (nt == NTILES - 1) __builtin_amdgcn_s_wait_tensorcnt(0);
            else                  __builtin_amdgcn_s_wait_tensorcnt(1);
        }
        __syncthreads();                       // tile nt visible to all waves
        const __bf16* tb = &BtC[nt & 1][0];
#else
        __syncthreads();                       // previous tile's LDS reads done
        {   // Synchronous fallback staging: 4 KB tile, one uint4 per thread.
            int o    = threadIdx.x * 16;       // byte offset in 4 KB data
            int brow = o >> 8;                 // 256 B per combined row
            int bcol = o & 255;
            const uint4* src = (const uint4*)((const char*)embC + (size_t)n0 * 256 + o);
            *(uint4*)((char*)&BtC[0][0] + brow * (CROW * 2) + bcol) = *src;
        }
        __syncthreads();
        const __bf16* tb = &BtC[0][0];
#endif

        // --- 6 x V_WMMA_F32_16X16X32_BF16 per 16x16 tile (two K=32 chunks).
        v8f acc0 = {};   // hi*hi chain
        v8f acc1 = {};   // cross-term chain
#pragma unroll
        for (int c = 0; c < 2; ++c) {
            // 16-bit B layout: lane l holds N=l%16, K = c*32 + {8*kh..+7, 16+8*kh..+7}.
            const __bf16* ph = tb + row * CROW + c * 32 + 8 * kh;         // hi half
            const __bf16* pl = ph + 64;                                   // lo half
            v8bf h0 = *(const v8bf*)ph, h1 = *(const v8bf*)(ph + 16);
            v8bf l0 = *(const v8bf*)pl, l1 = *(const v8bf*)(pl + 16);
            v16bf bhi = __builtin_shufflevector(h0, h1, 0,1,2,3,4,5,6,7,8,9,10,11,12,13,14,15);
            v16bf blo = __builtin_shufflevector(l0, l1, 0,1,2,3,4,5,6,7,8,9,10,11,12,13,14,15);
            acc0 = __builtin_amdgcn_wmma_f32_16x16x32_bf16(
                false, a_hi[c], false, bhi, (short)0, acc0, false, false);
            acc1 = __builtin_amdgcn_wmma_f32_16x16x32_bf16(
                false, a_hi[c], false, blo, (short)0, acc1, false, false);
            acc1 = __builtin_amdgcn_wmma_f32_16x16x32_bf16(
                false, a_lo[c], false, bhi, (short)0, acc1, false, false);
        }

        // score = ||e_n||^2 - 2*dot   (||z||^2 dropped: row-constant)
        const float en   = senorm[n0 + row];
        const int   ncol = n0 + row;
#pragma unroll
        for (int v = 0; v < 8; ++v) {
            float score = en - 2.0f * (acc0[v] + acc1[v]);  // C/D: M = v + 8*(l/16)
            if (score < best[v]) { best[v] = score; bidx[v] = ncol; }
        }

#if USE_TDM
        __syncthreads();                        // all waves done reading buf[nt&1]
        if (wave == 0 && nt + 2 < NTILES)
            tdm_load_tile(embC, (nt + 2) * 16, (unsigned)(size_t)&BtC[nt & 1][0]);
#endif
    }

    // --- Min-reduce across the 16 lanes sharing each row (butterfly stays in-half).
#pragma unroll
    for (int off = 1; off < 16; off <<= 1) {
#pragma unroll
        for (int v = 0; v < 8; ++v) {
            float ov = __shfl_xor(best[v], off, 32);
            int   oi = __shfl_xor(bidx[v], off, 32);
            if (ov < best[v] || (ov == best[v] && oi < bidx[v])) {
                best[v] = ov; bidx[v] = oi;
            }
        }
    }

    // lane 0 owns rows m0+0..7, lane 16 owns rows m0+8..15
    if (row == 0) {
        const int rbase = kh * 8;
#pragma unroll
        for (int v = 0; v < 8; ++v) {
            int id = bidx[v];
            sidx[wave * 16 + rbase + v] = id;
            idx_out[m0 + rbase + v]     = (float)id;
            __hip_atomic_fetch_add(count_sum + id, 1.0f,
                                   __ATOMIC_RELAXED, __HIP_MEMORY_SCOPE_AGENT);
        }
    }
    __syncthreads();

    // --- Gather z_q (coalesced) and segment-sum z_e into weight_sum (f32 atomics).
#pragma unroll 4
    for (int r = 0; r < 16; ++r) {
        const int    id = sidx[wave * 16 + r];
        const size_t m  = (size_t)(m0 + r);
        float e0 = emb[(size_t)id * DIM + lane];
        float e1 = emb[(size_t)id * DIM + 32 + lane];
        zq_out[m * DIM + lane]      = e0;
        zq_out[m * DIM + 32 + lane] = e1;
        float z0 = z_e[m * DIM + lane];
        float z1 = z_e[m * DIM + 32 + lane];
        __hip_atomic_fetch_add(weight_sum + (size_t)id * DIM + lane, z0,
                               __ATOMIC_RELAXED, __HIP_MEMORY_SCOPE_AGENT);
        __hip_atomic_fetch_add(weight_sum + (size_t)id * DIM + 32 + lane, z1,
                               __ATOMIC_RELAXED, __HIP_MEMORY_SCOPE_AGENT);
    }
}

// ---------------------------------------------------------------------------
// Kernel 2: EMA finalize (single block of 1024 threads, LDS+shuffle reduction).
// ---------------------------------------------------------------------------
__global__ __launch_bounds__(1024) void vq_finalize_kernel(const float* __restrict__ ema_count,
                                                           const float* __restrict__ ema_weight,
                                                           const float* __restrict__ count_sum,
                                                           const float* __restrict__ weight_sum,
                                                           float* __restrict__ emb_new,
                                                           float* __restrict__ count_new,
                                                           float* __restrict__ weight_new) {
    __shared__ float red[32];
    __shared__ float scount[N_EMB];
    const int t = threadIdx.x;

    float c = 0.99f * ema_count[t] + 0.01f * count_sum[t];

    float s = c;
#pragma unroll
    for (int off = 16; off >= 1; off >>= 1) s += __shfl_xor(s, off, 32);
    if ((t & 31) == 0) red[t >> 5] = s;
    __syncthreads();
    if (t < 32) {
        float v = red[t];
#pragma unroll
        for (int off = 16; off >= 1; off >>= 1) v += __shfl_xor(v, off, 32);
        if (t == 0) red[0] = v;
    }
    __syncthreads();
    const float total = red[0];

    float sm = (c + 1e-5f) / (total + (float)N_EMB * 1e-5f) * total;
    count_new[t] = sm;
    scount[t]    = sm;
    __syncthreads();

#pragma unroll
    for (int j = 0; j < DIM; ++j) {
        int i    = j * N_EMB + t;
        int rowk = i >> 6;
        float w  = 0.99f * ema_weight[i] + 0.01f * weight_sum[i];
        weight_new[i] = w;
        emb_new[i]    = w / scount[rowk];
    }
}

// ---------------------------------------------------------------------------
extern "C" void kernel_launch(void* const* d_in, const int* in_sizes, int n_in,
                              void* d_out, int out_size, void* d_ws, size_t ws_size,
                              hipStream_t stream) {
    (void)in_sizes; (void)n_in; (void)out_size; (void)ws_size;

    const float* z_e        = (const float*)d_in[0];   // [131072, 64]
    const float* emb        = (const float*)d_in[1];   // [1024, 64]
    const float* ema_count  = (const float*)d_in[2];   // [1024]
    const float* ema_weight = (const float*)d_in[3];   // [1024, 64]

    float* out     = (float*)d_out;
    float* zq      = out;                               // [131072*64]
    float* idxo    = zq + (size_t)N_ROWS * DIM;         // [131072] (float-valued)
    float* emb_new = idxo + N_ROWS;                     // [1024*64]
    float* cnt_new = emb_new + N_EMB * DIM;             // [1024]
    float* w_new   = cnt_new + N_EMB;                   // [1024*64]

    float* ws         = (float*)d_ws;
    float* count_sum  = ws;                             // [1024]
    float* weight_sum = ws + N_EMB;                     // [65536]
    float* enorm      = ws + N_EMB + N_EMB * DIM;       // [1024]
    unsigned short* embC = (unsigned short*)(ws + N_EMB + N_EMB * DIM + N_EMB);  // [1024*128]

    vq_init_kernel<<<(N_EMB * DIM + 255) / 256, 256, 0, stream>>>(emb, ws, embC);
    vq_main_kernel<<<N_ROWS / (16 * 8), 256, 0, stream>>>(z_e, emb, embC, enorm,
                                                          zq, idxo, count_sum, weight_sum);
    vq_finalize_kernel<<<1, 1024, 0, stream>>>(ema_count, ema_weight, count_sum,
                                               weight_sum, emb_new, cnt_new, w_new);
}
